// EmbedMatcher_60610578482011
// MI455X (gfx1250) — matrix operand order repaired and checked by reading, not compiled, not run on hardware
//
#include <hip/hip_runtime.h>

// ---------------------------------------------------------------------------
// EmbedMatcher for MI455X (gfx1250, wave32, WMMA f32_16x16x32_f16)
// Workspace requirement: ~82 MB.
// Key simplification: softmax over a single support row == 1, so the LSTM
// attention read-out r == support_g (constant). gates = q@Wih^T + h@Whh_h^T
// + cvec, with cvec = b_ih + b_hh + support_g@Whh_r^T precomputed.
// ---------------------------------------------------------------------------

typedef __attribute__((ext_vector_type(16))) _Float16 v16h;
typedef __attribute__((ext_vector_type(8)))  _Float16 h8v;
typedef __attribute__((ext_vector_type(8)))  float    v8f;
typedef __attribute__((ext_vector_type(4)))  float    f4v;

#define PAD_IDXC 100000
#define ED   256
#define DMOD 512
#define BQN  4096
#define BSN  5
#define KNB  50

__device__ __forceinline__ float sigmoidf_(float x){ return 1.0f/(1.0f + __expf(-x)); }

__device__ __forceinline__ v8f wmma_f16(v16h a, v16h b, v8f c){
  return __builtin_amdgcn_wmma_f32_16x16x32_f16(false, a, false, b, (short)0, c, false, false);
}

// A fragment (16x32, f16) from a row-major f32 matrix, converting on load.
// Lane l holds row m0+(l&15); elements 0..7 -> K = k0+8*(l>>4)+i,
// elements 8..15 -> K = k0+16+8*(l>>4)+(i-8).   (ISA 7.12.2)
__device__ __forceinline__ v16h frag_a_f32(const float* A, int ld, int m0, int k0, int lane){
  int m  = m0 + (lane & 15);
  int kb = k0 + ((lane >> 4) << 3);
  const float* p = A + (size_t)m * ld + kb;
  f4v x0 = *(const f4v*)(p);
  f4v x1 = *(const f4v*)(p + 4);
  f4v x2 = *(const f4v*)(p + 16);
  f4v x3 = *(const f4v*)(p + 20);
  v16h a;
  #pragma unroll
  for (int i = 0; i < 4; ++i){
    a[i]      = (_Float16)x0[i];
    a[4 + i]  = (_Float16)x1[i];
    a[8 + i]  = (_Float16)x2[i];
    a[12 + i] = (_Float16)x3[i];
  }
  return a;
}

// A fragment from a row-major f16 matrix (global or LDS).
__device__ __forceinline__ v16h frag_a_f16(const _Float16* A, int ld, int m0, int k0, int lane){
  int m  = m0 + (lane & 15);
  int kb = k0 + ((lane >> 4) << 3);
  const _Float16* p = A + (size_t)m * ld + kb;
  h8v lo = *(const h8v*)p;
  h8v hi = *(const h8v*)(p + 16);
  v16h a;
  #pragma unroll
  for (int i = 0; i < 8; ++i){ a[i] = lo[i]; a[8 + i] = hi[i]; }
  return a;
}

// B fragment (32x16): B[k][n] = W[n0+n][k0+k] with W row-major [N,K] f16.
// Lane l holds col n0+(l&15); elements 0..7 -> K = k0+8*(l>>4)+i,
// elements 8..15 -> K = k0+16+8*(l>>4)+(i-8).
__device__ __forceinline__ v16h frag_b_w(const _Float16* W, int ld, int n0, int k0, int lane){
  int n  = n0 + (lane & 15);
  int kb = k0 + ((lane >> 4) << 3);
  const _Float16* p = W + (size_t)n * ld + kb;
  h8v lo = *(const h8v*)p;
  h8v hi = *(const h8v*)(p + 16);
  v16h b;
  #pragma unroll
  for (int i = 0; i < 8; ++i){ b[i] = lo[i]; b[8 + i] = hi[i]; }
  return b;
}

// ---------------------------------------------------------------------------
__global__ void cvt_f32_to_f16(const float* __restrict__ s, _Float16* __restrict__ d, int n){
  int i = blockIdx.x * 256 + threadIdx.x;
  if (i < n) d[i] = (_Float16)s[i];
}

// --------------------------- neighbor encoder ------------------------------
// One workgroup (256 thr = 8 waves) per batch row b.
__global__ __launch_bounds__(256) void ne_kernel(
    const int* __restrict__ conn, const int* __restrict__ selfids, int side,
    const float* __restrict__ emb, const _Float16* __restrict__ projh,
    const float* __restrict__ proj_lb, const float* __restrict__ proj_b,
    const float* __restrict__ gate_w, const float* __restrict__ gate_lb,
    const float* __restrict__ gate_b, float* __restrict__ outv, int out_off)
{
  __shared__ __align__(16) _Float16 s_nv[64 * ED];   // 32 KB: padded nv tile
  __shared__ float s_col[ED];
  __shared__ float s_self[ED];
  __shared__ float s_red[ED];
  __shared__ float s_mask[64];
  __shared__ float s_cnt;

  const int b = blockIdx.x;
  const int t = threadIdx.x;
  const int lane = t & 31, wv = t >> 5;

  // Stage nv = emb[rel]*emb[ent] (f16), rows 50..63 zero for WMMA padding.
  for (int k = 0; k < 64; ++k){
    float v = 0.f;
    if (k < KNB){
      int rel = conn[(b * KNB + k) * 2 + 0];
      int ent = conn[(b * KNB + k) * 2 + 1];
      v = emb[(size_t)rel * ED + t] * emb[(size_t)ent * ED + t];
      if (t == 0) s_mask[k] = (rel != PAD_IDXC) ? 1.f : 0.f;
    } else if (t == 0) s_mask[k] = 0.f;
    s_nv[k * ED + t] = (_Float16)v;
  }
  s_self[t] = emb[(size_t)selfids[b * 2 + side] * ED + t];
  s_col[t] = 0.f;
  __syncthreads();

  // GEMM: [64x256] = nv[64x256] x projW^T, wave wv -> m-tile (wv&3),
  // n-tiles (wv>>2)*8 .. +7.  K = 256 -> 8 WMMA steps per n-tile.
  const int mt = wv & 3;
  const int ng = wv >> 2;
  v8f acc[8];
  #pragma unroll
  for (int f = 0; f < 8; ++f) acc[f] = (v8f){0,0,0,0,0,0,0,0};
  for (int kk = 0; kk < 8; ++kk){
    int k0 = kk * 32;
    v16h a = frag_a_f16(s_nv, ED, mt * 16, k0, lane);
    #pragma unroll
    for (int f = 0; f < 8; ++f){
      v16h bb = frag_b_w(projh, ED, (ng * 8 + f) * 16, k0, lane);
      acc[f] = wmma_f16(a, bb, acc[f]);
    }
  }

  // Epilogue: bias + leaky-relu(0.01) + masked column sums (pool over k).
  const int lhalf = lane >> 4;
  #pragma unroll
  for (int f = 0; f < 8; ++f){
    int n = (ng * 8 + f) * 16 + (lane & 15);
    float bias = proj_lb[n] + proj_b[n];
    float lsum = 0.f;
    #pragma unroll
    for (int v = 0; v < 8; ++v){
      int kr = mt * 16 + lhalf * 8 + v;
      float x = acc[f][v] + bias;
      x = x > 0.f ? x : 0.01f * x;
      lsum += x * s_mask[kr];
    }
    atomicAdd(&s_col[n], lsum);
  }
  __syncthreads();
  if (t == 0){
    float c = 0.f;
    for (int k = 0; k < KNB; ++k) c += s_mask[k];
    s_cnt = c;
  }
  __syncthreads();

  float agg = s_col[t] / (s_cnt + 1e-9f);
  s_col[t] = agg;
  s_red[t] = gate_w[t] * s_self[t] + gate_w[ED + t] * agg;
  __syncthreads();
  for (int s = 128; s > 0; s >>= 1){
    if (t < s) s_red[t] += s_red[t + s];
    __syncthreads();
  }
  float gate = sigmoidf_(s_red[0] + gate_lb[0] + gate_b[0]);
  outv[(size_t)b * DMOD + out_off + t] = tanhf(s_self[t] + gate * s_col[t]);
}

// --------------------------- SE FFN part 1: h1 = relu(x@w1^T+b1) ----------
__global__ __launch_bounds__(256) void se1_kernel(
    const float* __restrict__ x, const _Float16* __restrict__ w1h,
    const float* __restrict__ b1, _Float16* __restrict__ h1h)
{
  const int lane = threadIdx.x & 31, wv = threadIdx.x >> 5;
  const int mt = wv & 3, ng = wv >> 2;
  const int mbase = blockIdx.x * 64 + mt * 16;
  const int nblk  = blockIdx.y * 128;
  v8f acc[4];
  #pragma unroll
  for (int f = 0; f < 4; ++f) acc[f] = (v8f){0,0,0,0,0,0,0,0};
  for (int kk = 0; kk < 16; ++kk){
    v16h a = frag_a_f32(x, 512, mbase, kk * 32, lane);
    #pragma unroll
    for (int f = 0; f < 4; ++f){
      v16h bb = frag_b_w(w1h, 512, nblk + (ng * 4 + f) * 16, kk * 32, lane);
      acc[f] = wmma_f16(a, bb, acc[f]);
    }
  }
  const int lhalf = lane >> 4;
  #pragma unroll
  for (int f = 0; f < 4; ++f){
    int n = nblk + (ng * 4 + f) * 16 + (lane & 15);
    float bias = b1[n];
    #pragma unroll
    for (int v = 0; v < 8; ++v){
      int m = mbase + lhalf * 8 + v;
      float val = fmaxf(acc[f][v] + bias, 0.f);
      h1h[(size_t)m * 1024 + n] = (_Float16)val;
    }
  }
}

// ----------------- SE FFN part 2: h2 = h1@w2^T + b2 + x  ------------------
__global__ __launch_bounds__(256) void se2_kernel(
    const _Float16* __restrict__ h1h, const _Float16* __restrict__ w2h,
    const float* __restrict__ b2, const float* __restrict__ x,
    float* __restrict__ h2)
{
  const int lane = threadIdx.x & 31, wv = threadIdx.x >> 5;
  const int mt = wv & 3, ng = wv >> 2;
  const int mbase = blockIdx.x * 64 + mt * 16;
  const int nblk  = blockIdx.y * 128;
  v8f acc[4];
  #pragma unroll
  for (int f = 0; f < 4; ++f) acc[f] = (v8f){0,0,0,0,0,0,0,0};
  for (int kk = 0; kk < 32; ++kk){
    v16h a = frag_a_f16(h1h, 1024, mbase, kk * 32, lane);
    #pragma unroll
    for (int f = 0; f < 4; ++f){
      v16h bb = frag_b_w(w2h, 1024, nblk + (ng * 4 + f) * 16, kk * 32, lane);
      acc[f] = wmma_f16(a, bb, acc[f]);
    }
  }
  const int lhalf = lane >> 4;
  #pragma unroll
  for (int f = 0; f < 4; ++f){
    int n = nblk + (ng * 4 + f) * 16 + (lane & 15);
    float bias = b2[n];
    #pragma unroll
    for (int v = 0; v < 8; ++v){
      int m = mbase + lhalf * 8 + v;
      h2[(size_t)m * 512 + n] = acc[f][v] + bias + x[(size_t)m * 512 + n];
    }
  }
}

// ------------------------------ LayerNorm ---------------------------------
__global__ __launch_bounds__(256) void ln_kernel(
    const float* __restrict__ h2, const float* __restrict__ g,
    const float* __restrict__ bta, float* __restrict__ out)
{
  __shared__ float ssum[256], ssq[256];
  const int r = blockIdx.x, t = threadIdx.x;
  float a0 = h2[(size_t)r * 512 + t];
  float a1 = h2[(size_t)r * 512 + 256 + t];
  ssum[t] = a0 + a1;
  ssq[t]  = a0 * a0 + a1 * a1;
  __syncthreads();
  for (int s = 128; s > 0; s >>= 1){
    if (t < s){ ssum[t] += ssum[t + s]; ssq[t] += ssq[t + s]; }
    __syncthreads();
  }
  float mu  = ssum[0] * (1.f / 512.f);
  float var = ssq[0] * (1.f / 512.f) - mu * mu;
  float rs  = rsqrtf(var + 1e-5f);
  out[(size_t)r * 512 + t]       = g[t] * (a0 - mu) * rs + bta[t];
  out[(size_t)r * 512 + 256 + t] = g[256 + t] * (a1 - mu) * rs + bta[256 + t];
}

// ------------------- tiny support path (5 rows, scalar) -------------------
__global__ __launch_bounds__(256) void support_path_kernel(
    const float* __restrict__ sv, const float* __restrict__ w1, const float* __restrict__ b1,
    const float* __restrict__ w2, const float* __restrict__ b2,
    const float* __restrict__ g, const float* __restrict__ bb, float* __restrict__ supg)
{
  __shared__ float h1[BSN * 1024];
  __shared__ float hh[BSN * 512];
  __shared__ float mu[BSN], rs[BSN];
  const int t = threadIdx.x;
  for (int idx = t; idx < BSN * 1024; idx += 256){
    int r = idx >> 10, i = idx & 1023;
    float s = b1[i];
    for (int k = 0; k < 512; ++k) s += sv[r * 512 + k] * w1[(size_t)i * 512 + k];
    h1[idx] = fmaxf(s, 0.f);
  }
  __syncthreads();
  for (int idx = t; idx < BSN * 512; idx += 256){
    int r = idx >> 9, n = idx & 511;
    float s = b2[n] + sv[r * 512 + n];
    for (int k = 0; k < 1024; ++k) s += h1[r * 1024 + k] * w2[(size_t)n * 1024 + k];
    hh[idx] = s;
  }
  __syncthreads();
  if (t < BSN){
    float s = 0.f, q = 0.f;
    for (int n = 0; n < 512; ++n){ float v = hh[t * 512 + n]; s += v; q += v * v; }
    float m = s * (1.f / 512.f);
    mu[t] = m;
    rs[t] = rsqrtf(q * (1.f / 512.f) - m * m + 1e-5f);
  }
  __syncthreads();
  for (int n = t; n < 512; n += 256){
    float s = 0.f;
    for (int r = 0; r < BSN; ++r) s += g[n] * (hh[r * 512 + n] - mu[r]) * rs[r] + bb[n];
    supg[n] = s * (1.f / (float)BSN);
  }
}

// cvec[col] = b_ih+b_hh (+ support_g @ w_hh[col][512:])
__global__ void cvec_kernel(const float* __restrict__ b_ih, const float* __restrict__ b_hh,
                            const float* __restrict__ supg, const float* __restrict__ whh,
                            float* __restrict__ biassum, float* __restrict__ cvec)
{
  int col = blockIdx.x * 256 + threadIdx.x;
  float bs = b_ih[col] + b_hh[col];
  biassum[col] = bs;
  float s = bs;
  for (int k = 0; k < 512; ++k) s += supg[k] * whh[(size_t)col * 1024 + 512 + k];
  cvec[col] = s;
}

// ------------------------------ LSTM step ---------------------------------
// Each WG: rows mblk..+63, cols jblk..+63 in each of the 4 gate chunks.
// Cell update fused into the epilogue; gates never materialized.
__global__ __launch_bounds__(256) void lstm_step_kernel(
    const float* __restrict__ qenc, const float* __restrict__ hin,
    const _Float16* __restrict__ wihh, const _Float16* __restrict__ whhh,
    const float* __restrict__ cvec, float* __restrict__ c_ws,
    float* __restrict__ hout, int use_h)
{
  const int lane = threadIdx.x & 31, wv = threadIdx.x >> 5;
  const int mt = wv & 3, ng = wv >> 2;
  const int mbase = blockIdx.x * 64 + mt * 16;
  const int jbase = blockIdx.y * 64 + ng * 32;
  v8f acc[4][2];
  #pragma unroll
  for (int c = 0; c < 4; ++c)
    #pragma unroll
    for (int tf = 0; tf < 2; ++tf) acc[c][tf] = (v8f){0,0,0,0,0,0,0,0};

  for (int kk = 0; kk < 16; ++kk){           // query @ w_ih^T  (K = 512)
    v16h a = frag_a_f32(qenc, 512, mbase, kk * 32, lane);
    #pragma unroll
    for (int c = 0; c < 4; ++c)
      #pragma unroll
      for (int tf = 0; tf < 2; ++tf){
        v16h bb = frag_b_w(wihh, 512, c * 1024 + jbase + tf * 16, kk * 32, lane);
        acc[c][tf] = wmma_f16(a, bb, acc[c][tf]);
      }
  }
  if (use_h){                                 // h @ w_hh[:, :512]^T
    for (int kk = 0; kk < 16; ++kk){
      v16h a = frag_a_f32(hin, 512, mbase, kk * 32, lane);
      #pragma unroll
      for (int c = 0; c < 4; ++c)
        #pragma unroll
        for (int tf = 0; tf < 2; ++tf){
          v16h bb = frag_b_w(whhh, 1024, c * 1024 + jbase + tf * 16, kk * 32, lane);
          acc[c][tf] = wmma_f16(a, bb, acc[c][tf]);
        }
    }
  }

  const int lhalf = lane >> 4;
  #pragma unroll
  for (int tf = 0; tf < 2; ++tf){
    int j = jbase + tf * 16 + (lane & 15);
    float cv_i = cvec[j], cv_f = cvec[1024 + j], cv_g = cvec[2048 + j], cv_o = cvec[3072 + j];
    #pragma unroll
    for (int v = 0; v < 8; ++v){
      int m = mbase + lhalf * 8 + v;
      float iv = sigmoidf_(acc[0][tf][v] + cv_i);
      float fv = sigmoidf_(acc[1][tf][v] + cv_f);
      float gv = tanhf(acc[2][tf][v] + cv_g);
      float ov = sigmoidf_(acc[3][tf][v] + cv_o);
      float cold = use_h ? c_ws[(size_t)m * 1024 + j] : 0.f;
      float cnew = fv * cold + iv * gv;
      c_ws[(size_t)m * 1024 + j] = cnew;
      if (j < 512)
        hout[(size_t)m * 512 + j] = qenc[(size_t)m * 512 + j] + ov * tanhf(cnew);
    }
  }
}

// ------------------------------ final scores ------------------------------
__global__ __launch_bounds__(256) void dot_kernel(const float* __restrict__ h,
                                                  const float* __restrict__ sg,
                                                  float* __restrict__ out)
{
  const int wv = threadIdx.x >> 5, lane = threadIdx.x & 31;
  const int row = blockIdx.x * 8 + wv;
  float s = 0.f;
  for (int n = lane; n < 512; n += 32) s += h[(size_t)row * 512 + n] * sg[n];
  #pragma unroll
  for (int off = 16; off > 0; off >>= 1) s += __shfl_down(s, off, 32);
  if (lane == 0) out[row] = s;
}

// ---------------------------------------------------------------------------
extern "C" void kernel_launch(void* const* d_in, const int* in_sizes, int n_in,
                              void* d_out, int out_size, void* d_ws, size_t ws_size,
                              hipStream_t stream) {
  (void)in_sizes; (void)n_in; (void)out_size; (void)ws_size;

  const int*   query   = (const int*)  d_in[0];
  const int*   support = (const int*)  d_in[1];
  const int*   q_l1    = (const int*)  d_in[2];
  const int*   q_r1    = (const int*)  d_in[3];
  const int*   s_l1    = (const int*)  d_in[4];
  const int*   s_r1    = (const int*)  d_in[5];
  const float* emb     = (const float*)d_in[6];
  const float* proj_w  = (const float*)d_in[7];
  const float* proj_lb = (const float*)d_in[8];
  const float* proj_b  = (const float*)d_in[9];
  const float* gate_w  = (const float*)d_in[10];
  const float* gate_lb = (const float*)d_in[11];
  const float* gate_b  = (const float*)d_in[12];
  const float* se_w1   = (const float*)d_in[13];
  const float* se_b1   = (const float*)d_in[14];
  const float* se_w2   = (const float*)d_in[15];
  const float* se_b2   = (const float*)d_in[16];
  const float* ln_g    = (const float*)d_in[17];
  const float* ln_b    = (const float*)d_in[18];
  const float* w_ih    = (const float*)d_in[19];
  const float* w_hh    = (const float*)d_in[20];
  const float* b_ih    = (const float*)d_in[21];
  const float* b_hh    = (const float*)d_in[22];
  float* out = (float*)d_out;

  // ---- workspace bump allocator (256B aligned) ----
  char* wp = (char*)d_ws;
  auto alloc = [&](size_t bytes) -> void* {
    void* r = (void*)wp;
    wp += (bytes + 255) & ~(size_t)255;
    return r;
  };
  _Float16* projh = (_Float16*)alloc(256 * 256 * 2);
  _Float16* w1h   = (_Float16*)alloc(1024 * 512 * 2);
  _Float16* w2h   = (_Float16*)alloc(512 * 1024 * 2);
  _Float16* wihh  = (_Float16*)alloc(4096 * 512 * 2);
  _Float16* whhh  = (_Float16*)alloc(4096 * 1024 * 2);
  float* qvec  = (float*)alloc((size_t)BQN * 512 * 4);
  float* svec  = (float*)alloc((size_t)BSN * 512 * 4);
  _Float16* h1h = (_Float16*)alloc((size_t)BQN * 1024 * 2);
  float* h2    = (float*)alloc((size_t)BQN * 512 * 4);
  float* qenc  = (float*)alloc((size_t)BQN * 512 * 4);
  float* supg  = (float*)alloc(512 * 4);
  float* bsum  = (float*)alloc(4096 * 4);
  float* cvec  = (float*)alloc(4096 * 4);
  float* c_ws  = (float*)alloc((size_t)BQN * 1024 * 4);
  float* hA    = (float*)alloc((size_t)BQN * 512 * 4);
  float* hB    = (float*)alloc((size_t)BQN * 512 * 4);

  // 1) weight conversion to f16
  cvt_f32_to_f16<<<(65536 + 255) / 256, 256, 0, stream>>>(proj_w, projh, 65536);
  cvt_f32_to_f16<<<(524288 + 255) / 256, 256, 0, stream>>>(se_w1, w1h, 524288);
  cvt_f32_to_f16<<<(524288 + 255) / 256, 256, 0, stream>>>(se_w2, w2h, 524288);
  cvt_f32_to_f16<<<(2097152 + 255) / 256, 256, 0, stream>>>(w_ih, wihh, 2097152);
  cvt_f32_to_f16<<<(4194304 + 255) / 256, 256, 0, stream>>>(w_hh, whhh, 4194304);

  // 2) neighbor encoders
  ne_kernel<<<BQN, 256, 0, stream>>>(q_l1, query, 0, emb, projh, proj_lb, proj_b,
                                     gate_w, gate_lb, gate_b, qvec, 0);
  ne_kernel<<<BQN, 256, 0, stream>>>(q_r1, query, 1, emb, projh, proj_lb, proj_b,
                                     gate_w, gate_lb, gate_b, qvec, 256);
  ne_kernel<<<BSN, 256, 0, stream>>>(s_l1, support, 0, emb, projh, proj_lb, proj_b,
                                     gate_w, gate_lb, gate_b, svec, 0);
  ne_kernel<<<BSN, 256, 0, stream>>>(s_r1, support, 1, emb, projh, proj_lb, proj_b,
                                     gate_w, gate_lb, gate_b, svec, 256);

  // 3) support encoder on query batch
  se1_kernel<<<dim3(64, 8), 256, 0, stream>>>(qvec, w1h, se_b1, h1h);
  se2_kernel<<<dim3(64, 4), 256, 0, stream>>>(h1h, w2h, se_b2, qvec, h2);
  ln_kernel<<<BQN, 256, 0, stream>>>(h2, ln_g, ln_b, qenc);

  // 4) support path -> support_g, then cvec
  support_path_kernel<<<1, 256, 0, stream>>>(svec, se_w1, se_b1, se_w2, se_b2,
                                             ln_g, ln_b, supg);
  cvec_kernel<<<16, 256, 0, stream>>>(b_ih, b_hh, supg, w_hh, bsum, cvec);

  // 5) LSTM process block (4 steps, h double-buffered)
  lstm_step_kernel<<<dim3(64, 16), 256, 0, stream>>>(qenc, hB, wihh, whhh, bsum, c_ws, hA, 0);
  lstm_step_kernel<<<dim3(64, 16), 256, 0, stream>>>(qenc, hA, wihh, whhh, cvec, c_ws, hB, 1);
  lstm_step_kernel<<<dim3(64, 16), 256, 0, stream>>>(qenc, hB, wihh, whhh, cvec, c_ws, hA, 1);
  lstm_step_kernel<<<dim3(64, 16), 256, 0, stream>>>(qenc, hA, wihh, whhh, cvec, c_ws, hB, 1);

  // 6) scores
  dot_kernel<<<BQN / 8, 256, 0, stream>>>(hB, supg, out);
}
